// PVFC_61443802137152
// MI455X (gfx1250) — matrix operand order repaired and checked by reading, not compiled, not run on hardware
//
#include <hip/hip_runtime.h>

// ---------------------------------------------------------------------------
// PVFC (hyperbolic MLR + sinh activation) for MI455X / gfx1250, wave32 WMMA.
// GEMM via split-precision bf16 (hi/lo) with f32 accumulation:
//   x*z ~= xh*zh + xh*zl + xl*zh   (3x v_wmma_f32_16x16x32_bf16)
// ---------------------------------------------------------------------------

typedef __attribute__((ext_vector_type(16))) __bf16        v16bf;
typedef __attribute__((ext_vector_type(8)))  float         v8f;
typedef __attribute__((ext_vector_type(4)))  float         v4f;
typedef __attribute__((ext_vector_type(4)))  unsigned int  v4u;
typedef __attribute__((ext_vector_type(2)))  unsigned int  v2u;

#define B_DIM   16384
#define D_DIM   1024
#define OUT_DIM 4096

// workspace layout (float offsets)
#define WS_C1   0        // cosh(sr)/z_norm          [OUT]
#define WS_C2   4096     // sinh(sr)                 [OUT]
#define WS_C3   8192     // z_norm                   [OUT]
#define WS_C4   12288    // bias                     [OUT]
#define WS_BINV 16384    // sqrt(1 + ||x_row||^2)    [B]

// ---- bf16 helpers (RNE, bit tricks; no header-API dependence) -------------
__device__ __forceinline__ unsigned f2bf_bits(float f) {
    unsigned u = __builtin_bit_cast(unsigned, f);
    u += 0x7fffu + ((u >> 16) & 1u);     // round-to-nearest-even
    return u >> 16;
}
__device__ __forceinline__ float bfbits2f(unsigned s) {
    return __builtin_bit_cast(float, s << 16);
}

// convert 4 f32 -> packed bf16 hi (b64) + bf16 lo residual (b64), store to LDS
__device__ __forceinline__ void cvt_store4(const v4f a,
                                           unsigned char* hi_p,
                                           unsigned char* lo_p) {
    unsigned h0 = f2bf_bits(a.x), h1 = f2bf_bits(a.y);
    unsigned h2 = f2bf_bits(a.z), h3 = f2bf_bits(a.w);
    v2u hv = { h0 | (h1 << 16), h2 | (h3 << 16) };
    unsigned l0 = f2bf_bits(a.x - bfbits2f(h0));
    unsigned l1 = f2bf_bits(a.y - bfbits2f(h1));
    unsigned l2 = f2bf_bits(a.z - bfbits2f(h2));
    unsigned l3 = f2bf_bits(a.w - bfbits2f(h3));
    v2u lv = { l0 | (l1 << 16), l2 | (l3 << 16) };
    *(v2u*)hi_p = hv;
    *(v2u*)lo_p = lv;
}

union Frag { v4u q[2]; v16bf v; };

// ---------------------------------------------------------------------------
// Per-column constants: z_norm, cosh/sinh of clipped r, bias.
// ---------------------------------------------------------------------------
__global__ __launch_bounds__(256) void col_stats_kernel(
    const float* __restrict__ z, const float* __restrict__ r,
    const float* __restrict__ bias, float* __restrict__ ws)
{
    __shared__ float red[256];
    const int o = blockIdx.x, t = threadIdx.x;
    const float* zr = z + (size_t)o * D_DIM;
    float s = 0.f;
    for (int i = t; i < D_DIM; i += 256) { float v = zr[i]; s = fmaf(v, v, s); }
    red[t] = s;
    __syncthreads();
    for (int w = 128; w > 0; w >>= 1) {
        if (t < w) red[t] += red[t + w];
        __syncthreads();
    }
    if (t == 0) {
        float zn = sqrtf(red[0]);
        zn = fmaxf(zn, 1e-15f);
        float sr = fminf(fmaxf(r[o], -15.f), 15.f);
        ws[WS_C1 + o] = coshf(sr) / zn;
        ws[WS_C2 + o] = sinhf(sr);
        ws[WS_C3 + o] = zn;
        ws[WS_C4 + o] = bias[o];
    }
}

// ---------------------------------------------------------------------------
// Per-row constants: beta_inv = sqrt(1 + ||x||^2)
// ---------------------------------------------------------------------------
__global__ __launch_bounds__(256) void row_stats_kernel(
    const float* __restrict__ x, float* __restrict__ ws)
{
    __shared__ float red[256];
    const int b = blockIdx.x, t = threadIdx.x;
    const float* xr = x + (size_t)b * D_DIM;
    float s = 0.f;
    for (int i = t; i < D_DIM; i += 256) { float v = xr[i]; s = fmaf(v, v, s); }
    red[t] = s;
    __syncthreads();
    for (int w = 128; w > 0; w >>= 1) {
        if (t < w) red[t] += red[t + w];
        __syncthreads();
    }
    if (t == 0) ws[WS_BINV + b] = sqrtf(1.0f + red[0]);
}

// ---------------------------------------------------------------------------
// Main fused kernel: 128x128 tile per WG, 8 waves (2 M x 4 N), wave tile 64x32.
// LDS (single buffered, 40 KB): bf16 hi/lo planes for A(128x32) and B(128x32),
// row stride 80 B (16B-aligned rows; conflict-free b128 fragment reads).
// ---------------------------------------------------------------------------
__global__ __launch_bounds__(256) void pvfc_gemm_kernel(
    const float* __restrict__ x, const float* __restrict__ z,
    const float* __restrict__ ws, float* __restrict__ out)
{
    __shared__ __align__(16) unsigned char lds[40960];
    constexpr int AH = 0, AL = 10240, BH = 20480, BL = 30720;
    constexpr int LDB = 80; // bytes per LDS row (32 bf16 + 8 pad)

    const int tid  = threadIdx.x;
    const int bn   = blockIdx.x;    // N tile (OUT/128)
    const int bm   = blockIdx.y;    // M tile (B/128)
    const int wave = tid >> 5, lane = tid & 31;
    const int wm   = wave >> 2;     // 0..1 : wave row (64 rows each)
    const int wn   = wave & 3;      // 0..3 : wave col (32 cols each)
    const int ml   = lane & 15;
    const int hh   = lane >> 4;     // half-wave select

    v8f acc[4][2];
#pragma unroll
    for (int i = 0; i < 4; ++i)
#pragma unroll
        for (int j = 0; j < 2; ++j)
#pragma unroll
            for (int e = 0; e < 8; ++e) acc[i][j][e] = 0.0f;

    const float* xbase = x + (size_t)(bm * 128) * D_DIM;
    const float* zbase = z + (size_t)(bn * 128) * D_DIM;

    for (int k0 = 0; k0 < D_DIM; k0 += 32) {
        __syncthreads();   // previous iteration's fragment reads done
        // ---- load 128x32 f32 tiles of A (x) and B (z), coalesced float4,
        //      split to bf16 hi/lo and stage in LDS ----
#pragma unroll
        for (int i = 0; i < 4; ++i) {
            const int f   = tid + i * 256;       // float4 id, 0..1023
            const int row = f >> 3;              // 0..127
            const int c   = f & 7;               // float4 column
            v4f a = *(const v4f*)(xbase + (size_t)row * D_DIM + k0 + c * 4);
            v4f b = *(const v4f*)(zbase + (size_t)row * D_DIM + k0 + c * 4);
            const int off = row * LDB + c * 8;
            cvt_store4(a, &lds[AH + off], &lds[AL + off]);
            cvt_store4(b, &lds[BH + off], &lds[BL + off]);
        }
        __syncthreads();

        // ---- wave-level WMMA over the K=32 slab ----
#pragma unroll
        for (int ni = 0; ni < 2; ++ni) {
            const int rowB = wn * 32 + ni * 16 + ml;      // N index
            const int bb   = rowB * LDB + 32 * hh;        // K half select
            Frag bh, bl;
            bh.q[0] = *(const v4u*)&lds[BH + bb];
            bh.q[1] = *(const v4u*)&lds[BH + bb + 16];
            bl.q[0] = *(const v4u*)&lds[BL + bb];
            bl.q[1] = *(const v4u*)&lds[BL + bb + 16];
#pragma unroll
            for (int mi = 0; mi < 4; ++mi) {
                const int rowA = wm * 64 + mi * 16 + ml;  // M index
                const int ab   = rowA * LDB + 32 * hh;
                Frag ah, al;
                ah.q[0] = *(const v4u*)&lds[AH + ab];
                ah.q[1] = *(const v4u*)&lds[AH + ab + 16];
                al.q[0] = *(const v4u*)&lds[AL + ab];
                al.q[1] = *(const v4u*)&lds[AL + ab + 16];
                acc[mi][ni] = __builtin_amdgcn_wmma_f32_16x16x32_bf16(
                    false, ah.v, false, bh.v, (short)0, acc[mi][ni], false, false);
                acc[mi][ni] = __builtin_amdgcn_wmma_f32_16x16x32_bf16(
                    false, ah.v, false, bl.v, (short)0, acc[mi][ni], false, false);
                acc[mi][ni] = __builtin_amdgcn_wmma_f32_16x16x32_bf16(
                    false, al.v, false, bh.v, (short)0, acc[mi][ni], false, false);
            }
        }
    }

    // ---- epilogue: hyperbolic MLR + sinh activation, coalesced f32 stores --
    const float* c1   = ws + WS_C1;
    const float* c2   = ws + WS_C2;
    const float* c3   = ws + WS_C3;
    const float* c4   = ws + WS_C4;
    const float* binv = ws + WS_BINV;

#pragma unroll
    for (int ni = 0; ni < 2; ++ni) {
        const int colG = bn * 128 + wn * 32 + ni * 16 + ml;
        const float C1 = c1[colG];
        const float C2 = c2[colG];
        const float C3 = c3[colG];
        const float C4 = c4[colG];
#pragma unroll
        for (int mi = 0; mi < 4; ++mi) {
            const int row0 = bm * 128 + wm * 64 + mi * 16 + 8 * hh;
#pragma unroll
            for (int v = 0; v < 8; ++v) {
                const int row = row0 + v;
                const float bi  = binv[row];
                const float xz  = acc[mi][ni][v];
                float arg = C1 * xz - C2 * bi;            // (coshA - sinhB)/zn
                arg = fminf(fmaxf(arg, -1e6f), 1e6f);
                const float val = sinhf(fmaf(C3, asinhf(arg), C4));
                out[(size_t)row * OUT_DIM + colG] = val;
            }
        }
    }
}

// ---------------------------------------------------------------------------
extern "C" void kernel_launch(void* const* d_in, const int* in_sizes, int n_in,
                              void* d_out, int out_size, void* d_ws, size_t ws_size,
                              hipStream_t stream) {
    (void)in_sizes; (void)n_in; (void)out_size; (void)ws_size;
    const float* x    = (const float*)d_in[0];   // [16384, 1024]
    const float* z    = (const float*)d_in[1];   // [4096, 1024]
    const float* r    = (const float*)d_in[2];   // [4096, 1]
    const float* bias = (const float*)d_in[3];   // [4096]
    float* out = (float*)d_out;                  // [16384, 4096]
    float* ws  = (float*)d_ws;                   // 128 KB used

    col_stats_kernel<<<OUT_DIM, 256, 0, stream>>>(z, r, bias, ws);
    row_stats_kernel<<<B_DIM, 256, 0, stream>>>(x, ws);

    dim3 grid(OUT_DIM / 128, B_DIM / 128);       // (32, 128) = 4096 WGs
    pvfc_gemm_kernel<<<grid, dim3(256), 0, stream>>>(x, z, ws, out);
}
